// Sampler_17995912970511
// MI455X (gfx1250) — matrix operand order repaired and checked by reading, not compiled, not run on hardware
//
#include <hip/hip_runtime.h>
#include <hip/hip_bf16.h>

typedef __attribute__((ext_vector_type(16))) _Float16 v16h;
typedef __attribute__((ext_vector_type(8)))  float    v8f;

#define B_TOT 65536
#define L_SEQ 128
#define H1 20
#define H2 2
#define G1 (4*H1)   /* 80 gate columns, 5 N-tiles */
#define G2 (4*H2)   /* 8 gate columns, 1 N-tile (padded) */
#define NEG_SLOPE 0.01f
#define WAVES 4
#define ROWS_PER_WAVE 16
#define ROWS_PER_BLOCK (WAVES*ROWS_PER_WAVE)

// ---- transcendentals: prefer the CDNA5 hardware v_tanh_f32 ----
#if __has_builtin(__builtin_amdgcn_tanhf)
__device__ __forceinline__ float ftanh_fast(float x) {
  return __builtin_amdgcn_tanhf(x);            // 1 TRANS op
}
__device__ __forceinline__ float fsig(float x) {
  // sigmoid(x) = 0.5*tanh(0.5x) + 0.5  -> 1 TRANS + mul + fma
  return __builtin_fmaf(0.5f, __builtin_amdgcn_tanhf(0.5f * x), 0.5f);
}
#else
__device__ __forceinline__ float fsig(float x) {
  return __builtin_amdgcn_rcpf(1.0f + __expf(-x));
}
__device__ __forceinline__ float ftanh_fast(float x) {
  return 2.0f * fsig(2.0f * x) - 1.0f;
}
#endif
__device__ __forceinline__ float lrelu(float x) {
  return x > 0.0f ? x : NEG_SLOPE * x;
}

// Concatenated LSTM1 weight: K in [0,32): [x(2) | h1(20) | bias-slot | pad]
__device__ __forceinline__ float w1cat(const float* Wih1, const float* Whh1,
                                       const float* bih1, const float* bhh1,
                                       int K, int c) {
  if (K < 2)        return Wih1[c*2 + K];
  if (K < 2+H1)     return Whh1[c*H1 + (K-2)];
  if (K == 2+H1)    return bih1[c] + bhh1[c];
  return 0.0f;
}
// Concatenated LSTM2 weight: K in [0,64): [out1(40) | h2(2) | bias-slot | pad]
__device__ __forceinline__ float w2cat(const float* Wih2, const float* Whh2,
                                       const float* bih2, const float* bhh2,
                                       int K, int c) {
  if (K < 2*H1)       return Wih2[c*(2*H1) + K];
  if (K < 2*H1+H2)    return Whh2[c*H2 + (K-2*H1)];
  if (K == 2*H1+H2)   return bih2[c] + bhh2[c];
  return 0.0f;
}

__global__ __launch_bounds__(WAVES*32)
void vae_lstm_sampler(
    const float* __restrict__ mu,   const float* __restrict__ log_var,
    const float* __restrict__ eps0, const float* __restrict__ eps,
    const float* __restrict__ Wih1, const float* __restrict__ Whh1,
    const float* __restrict__ bih1, const float* __restrict__ bhh1,
    const float* __restrict__ Wih2, const float* __restrict__ Whh2,
    const float* __restrict__ bih2, const float* __restrict__ bhh2,
    float* __restrict__ out_mu, float* __restrict__ out_lv,
    float* __restrict__ out_s)
{
  // Per-wave LDS staging (intra-wave only -> no barriers needed; DS in-order).
  __shared__ __align__(16) float    act1[WAVES][16][G1];  // activated LSTM1 gates
  __shared__ __align__(16) _Float16 a1s [WAVES][16][32];  // LSTM1 A staging (f16)
  __shared__ __align__(16) _Float16 a2s [WAVES][16][64];  // LSTM2 A staging (f16)
  __shared__ __align__(16) float    act2[WAVES][16][G2];  // activated LSTM2 gates
  __shared__ __align__(16) float    c1s [WAVES][16][H1];  // c1 state
  __shared__ __align__(16) float    c2s [WAVES][16][H2];  // c2 state

  const int lane = threadIdx.x & 31;
  const int w    = threadIdx.x >> 5;
  const int n    = lane & 15;   // WMMA N column / A row
  const int hg   = lane >> 4;   // lane half-group
  const int rowbase = blockIdx.x * ROWS_PER_BLOCK + w * ROWS_PER_WAVE;

  // ---- Build register-resident WMMA B-fragments for the weights (once) ----
  // B layout (16-bit, 32x16): elems 0..7 -> K=8*hg+e, elems 8..15 ->
  // K=16+8*hg+(e-8); N = lane%16.
  v16h w1f[5];
  #pragma unroll
  for (int tt = 0; tt < 5; ++tt) {
    const int c = tt*16 + n;
    #pragma unroll
    for (int e = 0; e < 16; ++e) {
      const int K = (e < 8) ? (8*hg + e) : (16 + 8*hg + (e - 8));
      w1f[tt][e] = (_Float16)w1cat(Wih1, Whh1, bih1, bhh1, K, c);
    }
  }
  v16h w2f[2];
  #pragma unroll
  for (int s = 0; s < 2; ++s) {
    #pragma unroll
    for (int e = 0; e < 16; ++e) {
      const int K = s*32 + ((e < 8) ? (8*hg + e) : (16 + 8*hg + (e - 8)));
      w2f[s][e] = (n < G2) ? (_Float16)w2cat(Wih2, Whh2, bih2, bhh2, K, n)
                           : (_Float16)0.0f;
    }
  }

  // ---- Init state ----
  for (int i = lane; i < 16*H1; i += 32) c1s[w][i/H1][i%H1] = 0.0f;
  c2s[w][lane>>1][lane&1] = 0.0f;
  for (int i = lane; i < 16*32; i += 32) a1s[w][i>>5][i&31] = (_Float16)0.0f;
  for (int i = lane; i < 16*64; i += 32) a2s[w][i>>6][i&63] = (_Float16)0.0f;
  if (lane < 16) {
    const int rowg = rowbase + lane;
    const float x0 = eps0[rowg] * __expf(0.5f * log_var[(size_t)rowg*L_SEQ])
                   + mu[(size_t)rowg*L_SEQ];
    a1s[w][lane][0]  = (_Float16)x0;       // x component 0
    a1s[w][lane][1]  = (_Float16)0.0f;     // x component 1
    a1s[w][lane][22] = (_Float16)1.0f;     // bias slot
    a2s[w][lane][42] = (_Float16)1.0f;     // bias slot
  }
  asm volatile("s_wait_dscnt 0x0" ::: "memory");

  // ---- 128 sequential steps ----
  for (int t = 0; t < L_SEQ; ++t) {
    // A1 fragment: 16-bit A 16x32 layout; lane (M=n, group hg):
    // elems 0..7 -> K=8*hg.., elems 8..15 -> K=16+8*hg..
    union { v16h h; int4 q[2]; } a1;
    a1.q[0] = *(const int4*)&a1s[w][n][8*hg];
    a1.q[1] = *(const int4*)&a1s[w][n][16 + 8*hg];

    // LSTM1 gate GEMM: 5 N-tiles of 16x16x32 f16 WMMA
    v8f acc[5];
    #pragma unroll
    for (int tt = 0; tt < 5; ++tt) {
      v8f z = {};
      acc[tt] = __builtin_amdgcn_wmma_f32_16x16x32_f16(
          false, a1.h, false, w1f[tt], (short)0, z, false, false);
    }

    // Activation + scatter gates to LDS (D layout: M = r + 8*hg, N = n)
    #pragma unroll
    for (int tt = 0; tt < 5; ++tt) {
      const int col = tt*16 + n;                 // gate column in [0,80)
      const bool is_tanh = (col >= 2*H1) && (col < 3*H1);  // 'g' gate
      #pragma unroll
      for (int r = 0; r < 8; ++r) {
        const float gv = acc[tt][r];
        act1[w][r + 8*hg][col] = is_tanh ? ftanh_fast(gv) : fsig(gv);
      }
    }
    asm volatile("s_wait_dscnt 0x0" ::: "memory");

    // Elementwise LSTM1 combine: 16 rows x 20 units = 320 tasks, 10/lane
    #pragma unroll
    for (int it = 0; it < 10; ++it) {
      const int task = lane + 32*it;
      const int row = task / H1;
      const int j   = task % H1;
      const float ig = act1[w][row][j];
      const float fg = act1[w][row][H1 + j];
      const float gg = act1[w][row][2*H1 + j];
      const float og = act1[w][row][3*H1 + j];
      const float cn = fg * c1s[w][row][j] + ig * gg;
      const float hn = og * ftanh_fast(cn);
      c1s[w][row][j]      = cn;
      a1s[w][row][2 + j]  = (_Float16)hn;            // h1 for next step's A1
      a2s[w][row][j]      = (_Float16)lrelu(hn);     // out1[0:20]
      a2s[w][row][H1 + j] = (_Float16)lrelu(cn);     // out1[20:40]
    }
    asm volatile("s_wait_dscnt 0x0" ::: "memory");

    // LSTM2 gate GEMM: K=64 -> two chained 16x16x32 WMMAs
    union { v16h h; int4 q[2]; } a2lo, a2hi;
    a2lo.q[0] = *(const int4*)&a2s[w][n][8*hg];
    a2lo.q[1] = *(const int4*)&a2s[w][n][16 + 8*hg];
    a2hi.q[0] = *(const int4*)&a2s[w][n][32 + 8*hg];
    a2hi.q[1] = *(const int4*)&a2s[w][n][48 + 8*hg];
    v8f z2 = {};
    v8f acc2 = __builtin_amdgcn_wmma_f32_16x16x32_f16(
        false, a2lo.h, false, w2f[0], (short)0, z2, false, false);
    acc2 = __builtin_amdgcn_wmma_f32_16x16x32_f16(
        false, a2hi.h, false, w2f[1], (short)0, acc2, false, false);

    if (n < G2) {
      const bool is_tanh2 = (n >= 4) && (n < 6);   // 'g' gate cols 4,5
      #pragma unroll
      for (int r = 0; r < 8; ++r) {
        const float gv = acc2[r];
        act2[w][r + 8*hg][n] = is_tanh2 ? ftanh_fast(gv) : fsig(gv);
      }
    }
    asm volatile("s_wait_dscnt 0x0" ::: "memory");

    // LSTM2 combine + outputs: 16 rows x 2 units = 32 tasks, 1/lane
    {
      const int row = lane >> 1, j = lane & 1;
      const float ig = act2[w][row][j];
      const float fg = act2[w][row][2 + j];
      const float gg = act2[w][row][4 + j];
      const float og = act2[w][row][6 + j];
      const float cn = fg * c2s[w][row][j] + ig * gg;
      const float hn = og * ftanh_fast(cn);
      c2s[w][row][j] = cn;
      a2s[w][row][40 + j] = (_Float16)hn;          // h2 for next step

      const int rowg = rowbase + row;
      const float other = __shfl_xor(hn, 1, 32);   // swap mu_r <-> lv_r
      if (j == 0) {
        out_mu[(size_t)rowg*L_SEQ + t] = hn;       // mu_r
      } else {
        out_lv[(size_t)rowg*L_SEQ + t] = hn;       // lv_r
        out_s [(size_t)rowg*L_SEQ + t] =
            eps[(size_t)t*B_TOT + rowg] * __expf(0.5f * hn) + other;
      }
    }

    // Next step's input: x = [mu[:, t], t]
    if (lane < 16) {
      const int rowg = rowbase + lane;
      a1s[w][lane][0] = (_Float16)mu[(size_t)rowg*L_SEQ + t];
      a1s[w][lane][1] = (_Float16)(float)t;
    }
    asm volatile("s_wait_dscnt 0x0" ::: "memory");
  }
}

extern "C" void kernel_launch(void* const* d_in, const int* in_sizes, int n_in,
                              void* d_out, int out_size, void* d_ws, size_t ws_size,
                              hipStream_t stream) {
  const float* mu      = (const float*)d_in[0];
  const float* log_var = (const float*)d_in[1];
  const float* eps0    = (const float*)d_in[2];
  const float* eps     = (const float*)d_in[3];
  const float* Wih1    = (const float*)d_in[4];
  const float* Whh1    = (const float*)d_in[5];
  const float* bih1    = (const float*)d_in[6];
  const float* bhh1    = (const float*)d_in[7];
  const float* Wih2    = (const float*)d_in[8];
  const float* Whh2    = (const float*)d_in[9];
  const float* bih2    = (const float*)d_in[10];
  const float* bhh2    = (const float*)d_in[11];

  float* out    = (float*)d_out;
  float* out_mu = out;
  float* out_lv = out + (size_t)B_TOT * L_SEQ;
  float* out_s  = out + 2 * (size_t)B_TOT * L_SEQ;

  dim3 grid(B_TOT / ROWS_PER_BLOCK);   // 1024 blocks
  dim3 block(WAVES * 32);              // 128 threads = 4 waves
  hipLaunchKernelGGL(vae_lstm_sampler, grid, block, 0, stream,
                     mu, log_var, eps0, eps,
                     Wih1, Whh1, bih1, bhh1,
                     Wih2, Whh2, bih2, bhh2,
                     out_mu, out_lv, out_s);
}
